// MinkUNet_learned_90615220011117
// MI455X (gfx1250) — matrix operand order repaired and checked by reading, not compiled, not run on hardware
//
#include <hip/hip_runtime.h>
#include <hip/hip_bf16.h>

// ---------------------------------------------------------------------------
// Types for CDNA5 WMMA
// ---------------------------------------------------------------------------
typedef __bf16 bf16;
typedef __attribute__((ext_vector_type(16))) __bf16 v16bf;
typedef __attribute__((ext_vector_type(8)))  float  v8f;

#define CONV_MODE_TABLE 0   // map[N][K] neighbor table (subm ks=3, down ks=2s2)
#define CONV_MODE_UP    1   // transposed conv: gather par[r] iff kidx[r]==k
#define CONV_MODE_IDENT 2   // 1x1 conv: identity gather, K==1

// ---------------------------------------------------------------------------
// Elementwise helpers
// ---------------------------------------------------------------------------
__global__ void k_f32_to_bf16(const float* __restrict__ x, bf16* __restrict__ y, long n) {
    long i = (long)blockIdx.x * blockDim.x + threadIdx.x;
    if (i < n) y[i] = (bf16)x[i];
}

// transpose-convert weights: [K][Cin][Cout] f32 -> [K][Cout][Cin] bf16
__global__ void k_wcvt(const float* __restrict__ w, bf16* __restrict__ wt,
                       int K, int Cin, int Cout) {
    long i = (long)blockIdx.x * blockDim.x + threadIdx.x;
    if (i >= (long)K * Cin * Cout) return;
    int  co = (int)(i % Cout);
    long t  = i / Cout;
    int  ci = (int)(t % Cin);
    int  k  = (int)(t / Cin);
    wt[((long)k * Cout + co) * Cin + ci] = (bf16)w[i];
}

__global__ void k_zero_f32(float* __restrict__ p, int n) {
    int i = blockIdx.x * blockDim.x + threadIdx.x;
    if (i < n) p[i] = 0.f;
}

// ---------------------------------------------------------------------------
// Gather-GEMM sparse convolution, v_wmma_f32_16x16x32_bf16, software-pipelined
// with double-buffered async global->LDS staging.
//
// Block = 256 threads (8 waves) -> 32-row x 64-col output tile
//   wave = (waveR in 0..1) x (waveC in 0..3), each computes a 16x16 subtile.
// Per pipeline slice s = (tap k, 32-wide Cin chunk):
//   A tile 32x32 bf16 gathered via global_load_async_to_lds_b64 (8B/lane),
//     stored in plain K order (K0..31) so each lane's WMMA fragment
//     (lanes 0-15: K0-15, lanes 16-31: K16-31) is one contiguous 32B LDS read.
//   B tile 64x32 bf16 staged K-major (sB[co][ci]) from pre-transposed weights
//     via global_load_async_to_lds_b128 (16B/lane); plain ci order is already
//     fragment order for B (lanes 0-15: K0-15, lanes 16-31: K16-31).
// Pipeline: at slice s each wave issues slice s+1 into the other LDS buffer,
// then s_wait_asynccnt 2 (ASYNCcnt retires in order per wave -> slice-s data
// has landed while slice-s+1 stays in flight), barrier, 1 WMMA, barrier.
// ---------------------------------------------------------------------------
__global__ __launch_bounds__(256)
void k_gather_conv_wmma(const bf16* __restrict__ x, const bf16* __restrict__ wt,
                        const int* __restrict__ map, const int* __restrict__ kid,
                        float* __restrict__ out,
                        int Nout, int Cin, int Cout, int K, int mode)
{
    __shared__ bf16 sA[2][32][32];   // [row][ci_local], plain K order
    __shared__ bf16 sB[2][64][32];   // [co_local][ci_local]

    const int tid   = threadIdx.x;
    const int lane  = tid & 31;
    const int wave  = tid >> 5;
    const int waveR = wave >> 2;
    const int waveC = wave & 3;
    const int rowBase = blockIdx.x * 32;
    const int colTile = blockIdx.y * 64;

    const int half = lane >> 4;
    const int mn   = lane & 15;

    // A staging geometry: 8 B (4 bf16) per thread
    const int ar   = tid >> 3;                 // row 0..31
    const int ac   = (tid & 7) * 4;            // ci within slice (mult of 4)
    const int apos = ac;
    // B staging geometry: 16 B (8 bf16) per thread
    const int bco  = tid >> 2;                 // co_local 0..63
    const int bci  = (tid & 3) * 8;            // ci within slice (mult of 8)

    const int  arow  = rowBase + ar;           // gather row for this thread
    const int  gco   = colTile + bco;          // output channel for this thread
    const bool fastB = (Cin % 32) == 0;

    const int nC = (Cin + 31) >> 5;            // 32-wide Cin slices
    const int S  = K * nC;                     // total pipeline slices

    auto issue = [&](int s, int buf) {
        int k  = s / nC;
        int c0 = (s - k * nC) << 5;
        // ---- A tile: async gather (zero-fill missing neighbors) ----
        {
            int g = -1;
            if (arow < Nout) {
                if (mode == CONV_MODE_TABLE)    g = map[(long)arow * K + k];
                else if (mode == CONV_MODE_UP)  g = (kid[arow] == k) ? map[arow] : -1;
                else                            g = arow;
            }
            int cc = c0 + ac;
            unsigned aoff = (unsigned)(unsigned long long)(const void*)&sA[buf][ar][apos];
            if (g >= 0 && cc + 4 <= Cin) {
                const bf16* gp = x + (long)g * Cin + cc;
                asm volatile("global_load_async_to_lds_b64 %0, %1, off"
                             :: "v"(aoff), "v"((unsigned long long)gp) : "memory");
            } else {
                *(uint2*)&sA[buf][ar][apos] = make_uint2(0u, 0u);
            }
        }
        // ---- B tile ----
        if (fastB) {
            unsigned boff = (unsigned)(unsigned long long)(const void*)&sB[buf][bco][bci];
            if (gco < Cout) {
                const bf16* gp = wt + ((long)k * Cout + gco) * Cin + c0 + bci;
                asm volatile("global_load_async_to_lds_b128 %0, %1, off"
                             :: "v"(boff), "v"((unsigned long long)gp) : "memory");
            } else {
                *(uint4*)&sB[buf][bco][bci] = make_uint4(0u, 0u, 0u, 0u);
            }
        } else { // stem: Cin not a multiple of 32
            #pragma unroll
            for (int t = 0; t < 8; ++t) {
                int e   = tid * 8 + t;
                int co  = e >> 5, ci = e & 31;
                int oc  = colTile + co, icc = c0 + ci;
                bf16 v = (bf16)0.f;
                if (oc < Cout && icc < Cin)
                    v = wt[((long)k * Cout + oc) * Cin + icc];
                sB[buf][co][ci] = v;
            }
        }
    };

    v8f acc = {0.f, 0.f, 0.f, 0.f, 0.f, 0.f, 0.f, 0.f};

    issue(0, 0);                                // prologue
    for (int s = 0; s < S; ++s) {
        const int cur = s & 1;
        if (s + 1 < S) {
            issue(s + 1, cur ^ 1);              // prefetch next slice
            asm volatile("s_wait_asynccnt 0x2" ::: "memory");  // slice s landed
        } else {
            asm volatile("s_wait_asynccnt 0x0" ::: "memory");
        }
        __syncthreads();

        // ---- fragments: one contiguous 32B LDS read each ----
        v16bf a = *(const v16bf*)&sA[cur][waveR * 16 + mn][half * 16];
        v16bf b = *(const v16bf*)&sB[cur][waveC * 16 + mn][half * 16];
        acc = __builtin_amdgcn_wmma_f32_16x16x32_bf16(
                  false, a, false, b, (short)0, acc, false, false);

        __syncthreads();                        // protect buf[cur] before reuse
    }

    // ---- store D (VGPR i -> M = i + 8*half, N = lane&15) ----
    #pragma unroll
    for (int i = 0; i < 8; ++i) {
        int r  = rowBase + waveR * 16 + i + 8 * half;
        int co = colTile + waveC * 16 + mn;
        if (r < Nout && co < Cout) out[(long)r * Cout + co] = acc[i];
    }
}

// ---------------------------------------------------------------------------
// BatchNorm: two-pass batch statistics. Stats on raw conv output suffice,
// since y = x*ae+be is affine: mean(y)=ae*mu+be, var(y)=ae^2*var.
// ---------------------------------------------------------------------------
__global__ __launch_bounds__(256)
void k_bn_stats(const float* __restrict__ x, float* __restrict__ s1,
                float* __restrict__ s2, int N, int C)
{
    extern __shared__ float sm[];   // 2*C floats
    float* l1 = sm;
    float* l2 = sm + C;
    for (int c = threadIdx.x; c < 2 * C; c += blockDim.x) sm[c] = 0.f;
    __syncthreads();
    long total = (long)N * C;
    for (long i = (long)blockIdx.x * blockDim.x + threadIdx.x; i < total;
         i += (long)gridDim.x * blockDim.x) {
        int c = (int)(i % C);
        float v = x[i];
        atomicAdd(&l1[c], v);
        atomicAdd(&l2[c], v * v);
    }
    __syncthreads();
    for (int c = threadIdx.x; c < C; c += blockDim.x) {
        atomicAdd(&s1[c], l1[c]);
        atomicAdd(&s2[c], l2[c]);
    }
}

__global__ void k_bn_finalize(const float* __restrict__ s1, const float* __restrict__ s2,
                              const float* __restrict__ ae, const float* __restrict__ be,
                              const float* __restrict__ gamma, const float* __restrict__ beta,
                              float* __restrict__ a, float* __restrict__ b, int N, int C)
{
    int c = blockIdx.x * blockDim.x + threadIdx.x;
    if (c >= C) return;
    float inv  = 1.f / (float)N;
    float mean = s1[c] * inv;
    float var  = s2[c] * inv - mean * mean;
    float aec = ae[c], bec = be[c];
    float m2 = aec * mean + bec;
    float v2 = aec * aec * var;
    float rs = rsqrtf(v2 + 1e-5f);
    float g  = gamma[c];
    a[c] = aec * rs * g;
    b[c] = (bec - m2) * rs * g + beta[c];
}

// fused affine + optional residual + optional ReLU, writes bf16 activation
__global__ void k_bn_apply(const float* __restrict__ x, const float* __restrict__ a,
                           const float* __restrict__ b, const bf16* __restrict__ res,
                           bf16* __restrict__ y, int N, int C, int relu)
{
    long i = (long)blockIdx.x * blockDim.x + threadIdx.x;
    if (i >= (long)N * C) return;
    int c = (int)(i % C);
    float v = x[i] * a[c] + b[c];
    if (res) v += (float)res[i];
    if (relu) v = fmaxf(v, 0.f);
    y[i] = (bf16)v;
}

__global__ void k_concat(const bf16* __restrict__ x1, int C1,
                         const bf16* __restrict__ x2, int C2,
                         bf16* __restrict__ y, int N)
{
    int C = C1 + C2;
    long i = (long)blockIdx.x * blockDim.x + threadIdx.x;
    if (i >= (long)N * C) return;
    long r = i / C;
    int  c = (int)(i % C);
    y[i] = (c < C1) ? x1[r * C1 + c] : x2[r * C2 + (c - C1)];
}

__global__ void k_bias(const float* __restrict__ x, const float* __restrict__ bias,
                       float* __restrict__ y, int N, int C)
{
    long i = (long)blockIdx.x * blockDim.x + threadIdx.x;
    if (i >= (long)N * C) return;
    int c = (int)(i % C);
    y[i] = x[i] + bias[c];
}

// ---------------------------------------------------------------------------
// Host orchestration
// ---------------------------------------------------------------------------
extern "C" void kernel_launch(void* const* d_in, const int* in_sizes, int n_in,
                              void* d_out, int out_size, void* d_ws, size_t ws_size,
                              hipStream_t stream)
{
    (void)out_size; (void)ws_size;

    // ---- maps live at the tail of d_in: nbr0..4, then (down,par,kidx) x 4 ----
    const int* nbr[5]; int Nl[5];
    const int mb = n_in - 17;
    for (int l = 0; l < 5; ++l) {
        nbr[l] = (const int*)d_in[mb + l];
        Nl[l]  = in_sizes[mb + l] / 27;
    }
    const int *down[4], *par[4], *kidm[4];
    for (int l = 0; l < 4; ++l) {
        down[l] = (const int*)d_in[mb + 5 + 3 * l];
        par[l]  = (const int*)d_in[mb + 5 + 3 * l + 1];
        kidm[l] = (const int*)d_in[mb + 5 + 3 * l + 2];
    }

    // ---- bump allocator over workspace ----
    char*  base = (char*)d_ws;
    size_t off  = 0;
    auto alloc = [&](size_t bytes) -> void* {
        void* p = base + off;
        off = (off + bytes + 255) & ~(size_t)255;
        return p;
    };

    // ---- params consumed in setup_inputs() insertion order ----
    int ii = 1; // d_in[0] = feats
    auto nextf = [&]() -> const float* { return (const float*)d_in[ii++]; };
    struct BNP { const float *ae, *be, *gamma, *beta; };
    auto nextbn = [&]() -> BNP {
        BNP p; p.ae = nextf(); p.be = nextf(); p.gamma = nextf(); p.beta = nextf();
        return p;
    };

    auto cvt = [&](const float* src, long n) -> bf16* {
        bf16* dst = (bf16*)alloc((size_t)n * sizeof(bf16));
        k_f32_to_bf16<<<(int)((n + 255) / 256), 256, 0, stream>>>(src, dst, n);
        return dst;
    };

    auto conv = [&](const bf16* x, const float* w, int K, int Cin, int Cout,
                    const int* map, const int* kx, int Nout, int mode) -> float* {
        long wn = (long)K * Cin * Cout;
        bf16* wt = (bf16*)alloc((size_t)wn * sizeof(bf16));
        k_wcvt<<<(int)((wn + 255) / 256), 256, 0, stream>>>(w, wt, K, Cin, Cout);
        float* out = (float*)alloc((size_t)Nout * Cout * sizeof(float));
        dim3 grid((Nout + 31) / 32, (Cout + 63) / 64);
        k_gather_conv_wmma<<<grid, 256, 0, stream>>>(x, wt, map, kx, out,
                                                     Nout, Cin, Cout, K, mode);
        return out;
    };

    auto bnapply = [&](float* h, BNP p, int N, int C, const bf16* res, int relu) -> bf16* {
        float* s = (float*)alloc((size_t)4 * C * sizeof(float)); // s1,s2,a,b
        k_zero_f32<<<(2 * C + 255) / 256, 256, 0, stream>>>(s, 2 * C);
        long tot = (long)N * C;
        int blocks = (int)((tot + 255) / 256);
        if (blocks > 2048) blocks = 2048;
        k_bn_stats<<<blocks, 256, 2 * C * sizeof(float), stream>>>(h, s, s + C, N, C);
        k_bn_finalize<<<(C + 63) / 64, 64, 0, stream>>>(s, s + C, p.ae, p.be, p.gamma,
                                                        p.beta, s + 2 * C, s + 3 * C, N, C);
        bf16* y = (bf16*)alloc((size_t)N * C * sizeof(bf16));
        k_bn_apply<<<(int)((tot + 255) / 256), 256, 0, stream>>>(h, s + 2 * C, s + 3 * C,
                                                                 res, y, N, C, relu);
        return y;
    };

    auto subm = [&](const bf16* x, int Cin, int Cout, int lvl,
                    const float* w, BNP bn, const bf16* res, int relu) -> bf16* {
        float* h = conv(x, w, 27, Cin, Cout, nbr[lvl], nullptr, Nl[lvl], CONV_MODE_TABLE);
        return bnapply(h, bn, Nl[lvl], Cout, res, relu);
    };

    auto rb = [&](const bf16* x, int Cin, int Cout, int lvl) -> bf16* {
        const float* w0 = nextf(); BNP b0 = nextbn();
        const float* w2 = nextf(); BNP b1 = nextbn();
        bf16* h   = subm(x, Cin, Cout, lvl, w0, b0, nullptr, 1);
        float* h2 = conv(h, w2, 27, Cout, Cout, nbr[lvl], nullptr, Nl[lvl], CONV_MODE_TABLE);
        const bf16* ds = x;
        if (Cin != Cout) {
            const float* wds = nextf(); BNP bds = nextbn();
            float* d = conv(x, wds, 1, Cin, Cout, nullptr, nullptr, Nl[lvl], CONV_MODE_IDENT);
            ds = bnapply(d, bds, Nl[lvl], Cout, nullptr, 0);
        }
        return bnapply(h2, b1, Nl[lvl], Cout, ds, 1);
    };

    auto downblock = [&](const bf16* x, int Cin, int Cout, int lin) -> bf16* {
        const float* w = nextf(); BNP bn = nextbn();
        float* h = conv(x, w, 8, Cin, Cout, down[lin], nullptr, Nl[lin + 1], CONV_MODE_TABLE);
        return bnapply(h, bn, Nl[lin + 1], Cout, nullptr, 1);
    };

    auto upblock = [&](const bf16* x, int Cin, int Cout, int lfine) -> bf16* {
        const float* w = nextf(); BNP bn = nextbn();
        float* h = conv(x, w, 8, Cin, Cout, par[lfine], kidm[lfine], Nl[lfine], CONV_MODE_UP);
        return bnapply(h, bn, Nl[lfine], Cout, nullptr, 1);
    };

    auto cat = [&](const bf16* a, int Ca, const bf16* b, int Cb, int N) -> bf16* {
        bf16* y = (bf16*)alloc((size_t)N * (Ca + Cb) * sizeof(bf16));
        long tot = (long)N * (Ca + Cb);
        k_concat<<<(int)((tot + 255) / 256), 256, 0, stream>>>(a, Ca, b, Cb, y, N);
        return y;
    };

    const int cs[9] = {32, 32, 64, 128, 256, 256, 128, 96, 96};

    // ---- forward ----
    bf16* xin = cvt((const float*)d_in[0], (long)Nl[0] * 4);

    const float* w; BNP bn;
    w = nextf(); bn = nextbn(); bf16* x0 = subm(xin, 4,     cs[0], 0, w, bn, nullptr, 1);
    w = nextf(); bn = nextbn();        x0 = subm(x0,  cs[0], cs[0], 0, w, bn, nullptr, 1);

    bf16* x1 = downblock(x0, cs[0], cs[0], 0);
    x1 = rb(x1, cs[0], cs[1], 1); x1 = rb(x1, cs[1], cs[1], 1);
    bf16* x2 = downblock(x1, cs[1], cs[1], 1);
    x2 = rb(x2, cs[1], cs[2], 2); x2 = rb(x2, cs[2], cs[2], 2);
    bf16* x3 = downblock(x2, cs[2], cs[2], 2);
    x3 = rb(x3, cs[2], cs[3], 3); x3 = rb(x3, cs[3], cs[3], 3);
    bf16* x4 = downblock(x3, cs[3], cs[3], 3);
    x4 = rb(x4, cs[3], cs[4], 4); x4 = rb(x4, cs[4], cs[4], 4);

    bf16* y1 = upblock(x4, cs[4], cs[5], 3);
    y1 = cat(y1, cs[5], x3, cs[3], Nl[3]);
    y1 = rb(y1, cs[5] + cs[3], cs[5], 3); y1 = rb(y1, cs[5], cs[5], 3);

    bf16* y2 = upblock(y1, cs[5], cs[6], 2);
    y2 = cat(y2, cs[6], x2, cs[2], Nl[2]);
    y2 = rb(y2, cs[6] + cs[2], cs[6], 2); y2 = rb(y2, cs[6], cs[6], 2);

    bf16* y3 = upblock(y2, cs[6], cs[7], 1);
    y3 = cat(y3, cs[7], x1, cs[1], Nl[1]);
    y3 = rb(y3, cs[7] + cs[1], cs[7], 1); y3 = rb(y3, cs[7], cs[7], 1);

    bf16* y4 = upblock(y3, cs[7], cs[8], 0);
    y4 = cat(y4, cs[8], x0, cs[0], Nl[0]);
    y4 = rb(y4, cs[8] + cs[0], cs[8], 0); y4 = rb(y4, cs[8], cs[8], 0);

    // ---- classifier: 1x1 conv + bias ----
    const float* clw = nextf();
    const float* clb = nextf();
    float* logits = conv(y4, clw, 1, cs[8], 19, nullptr, nullptr, Nl[0], CONV_MODE_IDENT);
    long tot = (long)Nl[0] * 19;
    k_bias<<<(int)((tot + 255) / 256), 256, 0, stream>>>(logits, clb, (float*)d_out, Nl[0], 19);
}
